// LCEF_7954279432420
// MI455X (gfx1250) — matrix operand order repaired and checked by reading, not compiled, not run on hardware
//
#include <hip/hip_runtime.h>
#include <hip/hip_bf16.h>

// MI455X / gfx1250 implementation.
// Pipeline: bn_prep + wsplit -> pool_max -> se_gate -> qk_gemm (WMMA bf16x3,
// pre-split weights, writes split bf16 qT/kT) -> sim_gemm (WMMA bf16x3) ->
// softmax_aff (folds alpha*s, emits split bf16 aff2) -> out_gemm (WMMA bf16x3,
// +x residual).  Workspace requirement: ~65 MB.

typedef __attribute__((ext_vector_type(16))) __bf16 v16bf;
typedef __attribute__((ext_vector_type(8)))  __bf16 v8bf;
typedef __attribute__((ext_vector_type(4)))  __bf16 v4bf;
typedef __attribute__((ext_vector_type(2)))  __bf16 v2bf;
typedef __attribute__((ext_vector_type(8)))  float  v8f;

static constexpr int Bb = 32;    // batch
static constexpr int Cc = 256;   // channels
static constexpr int Nn = 4096;  // length
static constexpr int N8 = 512;   // N/8

// ---------- fp32 -> bf16 hi/lo split (round-to-nearest-even) ----------
__device__ __forceinline__ float bf16_rne_f(float f) {
  unsigned u = __float_as_uint(f);
  u = (u + 0x7FFFu + ((u >> 16) & 1u)) & 0xFFFF0000u;
  return __uint_as_float(u);
}
__device__ __forceinline__ __bf16 bf16_bits(float rounded) {
  unsigned short s = (unsigned short)(__float_as_uint(rounded) >> 16);
  return __builtin_bit_cast(__bf16, s);
}
struct bf2 { __bf16 hi; __bf16 lo; };
__device__ __forceinline__ bf2 split_bf16(float f) {
  float fh = bf16_rne_f(f);
  bf2 r;
  r.hi = bf16_bits(fh);
  r.lo = bf16_bits(bf16_rne_f(f - fh));
  return r;
}

#if __has_builtin(__builtin_amdgcn_cvt_pk_bf16_f32)
// Hardware packed converter: 2 fp32 -> packed bf16 (RNE); lo = residual pass.
__device__ __forceinline__ void split16(const float* t, v16bf& hi, v16bf& lo) {
#pragma unroll
  for (int e = 0; e < 16; e += 2) {
    v2bf h = __builtin_amdgcn_cvt_pk_bf16_f32(t[e], t[e + 1]);
    float r0 = t[e] - (float)h[0];
    float r1 = t[e + 1] - (float)h[1];
    v2bf l = __builtin_amdgcn_cvt_pk_bf16_f32(r0, r1);
    hi[e] = h[0]; hi[e + 1] = h[1];
    lo[e] = l[0]; lo[e + 1] = l[1];
  }
}
#else
__device__ __forceinline__ void split16(const float* t, v16bf& hi, v16bf& lo) {
#pragma unroll
  for (int e = 0; e < 16; ++e) {
    bf2 r = split_bf16(t[e]);
    hi[e] = r.hi;
    lo[e] = r.lo;
  }
}
#endif

// assemble v16bf from two contiguous 8-element bf16 chunks (contiguous K: element e holds k = 16*kh + e)
__device__ __forceinline__ void frag_from2(const __bf16* p0, const __bf16* p1, v16bf& f) {
  v8bf a = *(const v8bf*)p0, b = *(const v8bf*)p1;
#pragma unroll
  for (int e = 0; e < 8; ++e) { f[e] = a[e]; f[8 + e] = b[e]; }
}

__device__ __forceinline__ v8f wmma_bf16(v16bf a, v16bf b, v8f c) {
  // D = A(16x32 bf16) * B(32x16 bf16) + C(16x16 f32)
  return __builtin_amdgcn_wmma_f32_16x16x32_bf16(false, a, false, b,
                                                 (short)0, c, false, false);
}
__device__ __forceinline__ void mma3(v8f& acc, const v16bf& ahi, const v16bf& alo,
                                     const v16bf& bhi, const v16bf& blo) {
  acc = wmma_bf16(ahi, bhi, acc);   // hi*hi
  acc = wmma_bf16(ahi, blo, acc);   // hi*lo
  acc = wmma_bf16(alo, bhi, acc);   // lo*hi  (lo*lo dropped: ~1.5e-5 rel)
}

// ---------- BN fold: scale = g*rsqrt(v+eps), shift = b - m*scale ----------
__global__ void bn_prep_kernel(const float* __restrict__ g1, const float* __restrict__ b1,
                               const float* __restrict__ m1, const float* __restrict__ v1,
                               const float* __restrict__ g2, const float* __restrict__ b2,
                               const float* __restrict__ m2, const float* __restrict__ v2,
                               float* __restrict__ sc1, float* __restrict__ sh1,
                               float* __restrict__ sc2, float* __restrict__ sh2) {
  int i = blockIdx.x * blockDim.x + threadIdx.x;
  if (i < N8) {
    float s1 = g1[i] * rsqrtf(v1[i] + 1e-5f);
    sc1[i] = s1; sh1[i] = b1[i] - m1[i] * s1;
    float s2 = g2[i] * rsqrtf(v2[i] + 1e-5f);
    sc2[i] = s2; sh2[i] = b2[i] - m2[i] * s2;
  }
}

// ---------- pre-split Wq|Wk into bf16 hi/lo (q at 0, k at N8*Nn) ----------
__global__ void wsplit_kernel(const float* __restrict__ Wq, const float* __restrict__ Wk,
                              __bf16* __restrict__ Whi, __bf16* __restrict__ Wlo) {
  size_t t = (size_t)blockIdx.x * blockDim.x + threadIdx.x;  // one float4 each
  const size_t half = (size_t)N8 * Nn / 4;
  const float4* src = (t < half) ? (const float4*)Wq : (const float4*)Wk;
  size_t idx = (t < half) ? t : t - half;
  float4 f = src[idx];
  bf2 s0 = split_bf16(f.x), s1 = split_bf16(f.y), s2 = split_bf16(f.z), s3 = split_bf16(f.w);
  v4bf h, l;
  h[0] = s0.hi; h[1] = s1.hi; h[2] = s2.hi; h[3] = s3.hi;
  l[0] = s0.lo; l[1] = s1.lo; l[2] = s2.lo; l[3] = s3.lo;
  ((v4bf*)Whi)[t] = h;
  ((v4bf*)Wlo)[t] = l;
}

// ---------- pooled[b,c] = max_n x[b,c,n] : one wave per row ----------
__global__ __launch_bounds__(32) void pool_max_kernel(const float* __restrict__ x,
                                                      float* __restrict__ pooled) {
  int bc = blockIdx.x;
  const float4* p = (const float4*)(x + (size_t)bc * Nn);
  int lane = threadIdx.x;
  float m = -3.402823466e38f;
#pragma unroll 4
  for (int j = 0; j < 32; ++j) {
    float4 v = p[lane + j * 32];
    m = fmaxf(m, fmaxf(fmaxf(v.x, v.y), fmaxf(v.z, v.w)));
  }
#pragma unroll
  for (int off = 16; off > 0; off >>= 1) m = fmaxf(m, __shfl_xor(m, off, 32));
  if (lane == 0) pooled[bc] = m;
}

// ---------- s[b,c] = sigmoid(relu(pooled @ W1^T) @ W2^T) ----------
__global__ void se_gate_kernel(const float* __restrict__ pooled,
                               const float* __restrict__ W1, const float* __restrict__ W2,
                               float* __restrict__ s) {
  __shared__ float ph[Cc];
  __shared__ float h[32];
  int b = blockIdx.x, t = threadIdx.x;
  ph[t] = pooled[b * Cc + t];
  __syncthreads();
  if (t < 32) {
    float a = 0.f;
#pragma unroll 8
    for (int i = 0; i < Cc; ++i) a += W1[t * Cc + i] * ph[i];
    h[t] = fmaxf(a, 0.f);
  }
  __syncthreads();
  float a = 0.f;
#pragma unroll
  for (int j = 0; j < 32; ++j) a += W2[t * 32 + j] * h[j];
  s[b * Cc + t] = 1.f / (1.f + expf(-a));
}

// ---------- q/k: relu(bn(W @ x[b]^T)) stored transposed + bf16-split ----------
// grid: (N8/64, C/32, B*2); 1 wave per 64x32 output tile; K = 4096
__global__ __launch_bounds__(32) void qk_gemm_kernel(
    const float* __restrict__ x,
    const __bf16* __restrict__ Whi, const __bf16* __restrict__ Wlo,
    const float* __restrict__ sc1, const float* __restrict__ sh1,
    const float* __restrict__ sc2, const float* __restrict__ sh2,
    __bf16* __restrict__ qT_hi, __bf16* __restrict__ qT_lo,
    __bf16* __restrict__ kT_hi, __bf16* __restrict__ kT_lo) {
  const int lane = threadIdx.x, l15 = lane & 15, kh = lane >> 4;
  const int i0 = blockIdx.x * 64;          // over N8 (4 M-tiles)
  const int c0 = blockIdx.y * 32;          // over C  (2 N-tiles)
  const int b = blockIdx.z >> 1, which = blockIdx.z & 1;
  const size_t wOff = (size_t)which * N8 * Nn;
  const __bf16* WHi = Whi + wOff;
  const __bf16* WLo = Wlo + wOff;
  const float* scale = which ? sc2 : sc1;
  const float* shift = which ? sh2 : sh1;
  __bf16* oHi = which ? kT_hi : qT_hi;
  __bf16* oLo = which ? kT_lo : qT_lo;
  const float* xb = x + (size_t)b * Cc * Nn;

  v8f acc[4][2] = {};
  for (int kk = 0; kk < Nn; kk += 32) {
    v16bf ahi[4], alo[4], bhi[2], blo[2];
#pragma unroll
    for (int mt = 0; mt < 4; ++mt) {   // A: pre-split W rows, contiguous K
      size_t off = (size_t)(i0 + 16 * mt + l15) * Nn + kk + 16 * kh;
      frag_from2(WHi + off, WHi + off + 8, ahi[mt]);
      frag_from2(WLo + off, WLo + off + 8, alo[mt]);
    }
#pragma unroll
    for (int nt = 0; nt < 2; ++nt) {   // B: x rows (N=c), K = kk+16*kh+e
      const float* p = xb + (size_t)(c0 + 16 * nt + l15) * Nn + kk + 16 * kh;
      __builtin_prefetch(p + 32, 0, 1);          // next K-step (global_prefetch_b8)
      const float4* p4 = (const float4*)p;
      float4 f0 = p4[0], f1 = p4[1], f2 = p4[2], f3 = p4[3];
      float t[16] = {f0.x, f0.y, f0.z, f0.w, f1.x, f1.y, f1.z, f1.w,
                     f2.x, f2.y, f2.z, f2.w, f3.x, f3.y, f3.z, f3.w};
      split16(t, bhi[nt], blo[nt]);
    }
#pragma unroll
    for (int mt = 0; mt < 4; ++mt)
#pragma unroll
      for (int nt = 0; nt < 2; ++nt) mma3(acc[mt][nt], ahi[mt], alo[mt], bhi[nt], blo[nt]);
  }
  // epilogue: BN + ReLU, store transposed qT[b][c][i] as hi/lo bf16
#pragma unroll
  for (int mt = 0; mt < 4; ++mt)
#pragma unroll
    for (int nt = 0; nt < 2; ++nt) {
      int ibase = i0 + 16 * mt + 8 * kh;
      v8bf h8, l8;
#pragma unroll
      for (int r = 0; r < 8; ++r) {
        float v = acc[mt][nt][r] * scale[ibase + r] + shift[ibase + r];
        v = fmaxf(v, 0.f);
        bf2 sp = split_bf16(v);
        h8[r] = sp.hi;
        l8[r] = sp.lo;
      }
      int c = c0 + 16 * nt + l15;
      size_t off = ((size_t)b * Cc + c) * N8 + ibase;
      *(v8bf*)(oHi + off) = h8;
      *(v8bf*)(oLo + off) = l8;
    }
}

// ---------- sim[b,c,d] = sum_i kT[b,c,i]*qT[b,d,i] ; K = 512, bf16 inputs ----------
__global__ __launch_bounds__(32) void sim_gemm_kernel(
    const __bf16* __restrict__ kT_hi, const __bf16* __restrict__ kT_lo,
    const __bf16* __restrict__ qT_hi, const __bf16* __restrict__ qT_lo,
    float* __restrict__ sim) {
  const int lane = threadIdx.x, l15 = lane & 15, kh = lane >> 4;
  const int c0 = blockIdx.x * 32, d0 = blockIdx.y * 32, b = blockIdx.z;
  v8f acc[2][2] = {};
  for (int kk = 0; kk < N8; kk += 32) {
    v16bf ahi[2], alo[2], bhi[2], blo[2];
#pragma unroll
    for (int mt = 0; mt < 2; ++mt) {
      size_t off = ((size_t)b * Cc + (c0 + 16 * mt + l15)) * N8 + kk + 16 * kh;
      frag_from2(kT_hi + off, kT_hi + off + 8, ahi[mt]);
      frag_from2(kT_lo + off, kT_lo + off + 8, alo[mt]);
    }
#pragma unroll
    for (int nt = 0; nt < 2; ++nt) {
      size_t off = ((size_t)b * Cc + (d0 + 16 * nt + l15)) * N8 + kk + 16 * kh;
      frag_from2(qT_hi + off, qT_hi + off + 8, bhi[nt]);
      frag_from2(qT_lo + off, qT_lo + off + 8, blo[nt]);
    }
#pragma unroll
    for (int mt = 0; mt < 2; ++mt)
#pragma unroll
      for (int nt = 0; nt < 2; ++nt) mma3(acc[mt][nt], ahi[mt], alo[mt], bhi[nt], blo[nt]);
  }
#pragma unroll
  for (int mt = 0; mt < 2; ++mt)
#pragma unroll
    for (int nt = 0; nt < 2; ++nt)
#pragma unroll
      for (int r = 0; r < 8; ++r) {
        int c = c0 + 16 * mt + 8 * kh + r, d = d0 + 16 * nt + l15;
        sim[((size_t)b * Cc + c) * Cc + d] = acc[mt][nt][r];
      }
}

// ---------- aff2 = alpha * s[b,d] * softmax_d(-sim[b,c,:]) -> split bf16 ----------
__global__ void softmax_aff_kernel(const float* __restrict__ sim_in,
                                   __bf16* __restrict__ aff_hi, __bf16* __restrict__ aff_lo,
                                   const float* __restrict__ s, const float* __restrict__ alpha) {
  __shared__ float red[Cc];
  int c = blockIdx.x, b = blockIdx.y, d = threadIdx.x;
  size_t idx = ((size_t)b * Cc + c) * Cc + d;
  float v = sim_in[idx];
  red[d] = v; __syncthreads();
#pragma unroll
  for (int off = 128; off > 0; off >>= 1) {
    if (d < off) red[d] = fminf(red[d], red[d + off]);
    __syncthreads();
  }
  float mn = red[0];
  __syncthreads();
  float e = expf(mn - v);                 // == stable softmax(max_row - sim)
  red[d] = e; __syncthreads();
#pragma unroll
  for (int off = 128; off > 0; off >>= 1) {
    if (d < off) red[d] += red[d + off];
    __syncthreads();
  }
  float val = alpha[0] * s[b * Cc + d] * e / red[0];
  bf2 sp = split_bf16(val);
  aff_hi[idx] = sp.hi;
  aff_lo[idx] = sp.lo;
}

// ---------- out[b,c,n] = sum_d aff2[b,c,d]*x[b,d,n] + x[b,c,n] ; K = 256 ----------
// grid: (C/64, Nn/32, B); 1 wave per 64x32 output tile
__global__ __launch_bounds__(32) void out_gemm_kernel(
    const __bf16* __restrict__ aff_hi, const __bf16* __restrict__ aff_lo,
    const float* __restrict__ x, float* __restrict__ out) {
  const int lane = threadIdx.x, l15 = lane & 15, kh = lane >> 4;
  const int c0 = blockIdx.x * 64, n0 = blockIdx.y * 32, b = blockIdx.z;
  v8f acc[4][2] = {};
  for (int kk = 0; kk < Cc; kk += 32) {
    v16bf ahi[4], alo[4], bhi[2], blo[2];
#pragma unroll
    for (int mt = 0; mt < 4; ++mt) {   // A: pre-split aff2 rows, contiguous in d
      size_t off = ((size_t)b * Cc + (c0 + 16 * mt + l15)) * Cc + kk + 16 * kh;
      frag_from2(aff_hi + off, aff_hi + off + 8, ahi[mt]);
      frag_from2(aff_lo + off, aff_lo + off + 8, alo[mt]);
    }
#pragma unroll
    for (int nt = 0; nt < 2; ++nt) {   // B: x columns, d-strided (L2-resident)
      int ncol = n0 + 16 * nt + l15;
      const float* p = x + ((size_t)b * Cc + (kk + 16 * kh)) * Nn + ncol;
      float t[16];
#pragma unroll
      for (int e = 0; e < 16; ++e) t[e] = p[(size_t)e * Nn];  // clause-able gathers
      split16(t, bhi[nt], blo[nt]);
    }
#pragma unroll
    for (int mt = 0; mt < 4; ++mt)
#pragma unroll
      for (int nt = 0; nt < 2; ++nt) mma3(acc[mt][nt], ahi[mt], alo[mt], bhi[nt], blo[nt]);
  }
#pragma unroll
  for (int mt = 0; mt < 4; ++mt)
#pragma unroll
    for (int nt = 0; nt < 2; ++nt)
#pragma unroll
      for (int r = 0; r < 8; ++r) {
        int c = c0 + 16 * mt + 8 * kh + r, n = n0 + 16 * nt + l15;
        size_t idx = ((size_t)b * Cc + c) * Nn + n;
        out[idx] = acc[mt][nt][r] + x[idx];
      }
}

extern "C" void kernel_launch(void* const* d_in, const int* in_sizes, int n_in,
                              void* d_out, int out_size, void* d_ws, size_t ws_size,
                              hipStream_t stream) {
  (void)in_sizes; (void)n_in; (void)out_size; (void)ws_size;
  const float* x  = (const float*)d_in[0];
  const float* Wq = (const float*)d_in[1];
  const float* g1 = (const float*)d_in[2];
  const float* b1 = (const float*)d_in[3];
  const float* m1 = (const float*)d_in[4];
  const float* v1 = (const float*)d_in[5];
  const float* Wk = (const float*)d_in[6];
  const float* g2 = (const float*)d_in[7];
  const float* b2 = (const float*)d_in[8];
  const float* m2 = (const float*)d_in[9];
  const float* v2 = (const float*)d_in[10];
  const float* W1 = (const float*)d_in[11];
  const float* W2 = (const float*)d_in[12];
  const float* alpha = (const float*)d_in[13];
  float* out = (float*)d_out;

  // ---- workspace carve (large, aligned chunks first) : ~65 MB ----
  char* ws = (char*)d_ws;
  const size_t qkElems = (size_t)Bb * Cc * N8;            // 4,194,304
  const size_t wElems  = (size_t)2 * N8 * Nn;             // 4,194,304 (Wq|Wk)
  const size_t afElems = (size_t)Bb * Cc * Cc;            // 2,097,152
  __bf16* qT_hi = (__bf16*)ws; ws += qkElems * sizeof(__bf16);
  __bf16* qT_lo = (__bf16*)ws; ws += qkElems * sizeof(__bf16);
  __bf16* kT_hi = (__bf16*)ws; ws += qkElems * sizeof(__bf16);
  __bf16* kT_lo = (__bf16*)ws; ws += qkElems * sizeof(__bf16);
  __bf16* Whi   = (__bf16*)ws; ws += wElems * sizeof(__bf16);
  __bf16* Wlo   = (__bf16*)ws; ws += wElems * sizeof(__bf16);
  float* sim    = (float*)ws;  ws += afElems * sizeof(float);
  __bf16* affHi = (__bf16*)ws; ws += afElems * sizeof(__bf16);
  __bf16* affLo = (__bf16*)ws; ws += afElems * sizeof(__bf16);
  float* pooled = (float*)ws;  ws += (size_t)Bb * Cc * sizeof(float);
  float* sgate  = (float*)ws;  ws += (size_t)Bb * Cc * sizeof(float);
  float* sc1 = (float*)ws; ws += N8 * sizeof(float);
  float* sh1 = (float*)ws; ws += N8 * sizeof(float);
  float* sc2 = (float*)ws; ws += N8 * sizeof(float);
  float* sh2 = (float*)ws; ws += N8 * sizeof(float);

  bn_prep_kernel<<<2, 256, 0, stream>>>(g1, b1, m1, v1, g2, b2, m2, v2, sc1, sh1, sc2, sh2);
  wsplit_kernel<<<(unsigned)(wElems / 4 / 256), 256, 0, stream>>>(Wq, Wk, Whi, Wlo);
  pool_max_kernel<<<Bb * Cc, 32, 0, stream>>>(x, pooled);
  se_gate_kernel<<<Bb, 256, 0, stream>>>(pooled, W1, W2, sgate);
  qk_gemm_kernel<<<dim3(N8 / 64, Cc / 32, Bb * 2), 32, 0, stream>>>(
      x, Whi, Wlo, sc1, sh1, sc2, sh2, qT_hi, qT_lo, kT_hi, kT_lo);
  sim_gemm_kernel<<<dim3(Cc / 32, Cc / 32, Bb), 32, 0, stream>>>(
      kT_hi, kT_lo, qT_hi, qT_lo, sim);
  softmax_aff_kernel<<<dim3(Cc, Bb), Cc, 0, stream>>>(sim, affHi, affLo, sgate, alpha);
  out_gemm_kernel<<<dim3(Cc / 64, Nn / 32, Bb), 32, 0, stream>>>(affHi, affLo, x, out);
}
